// HumanNodeEmbed_68109591380151
// MI455X (gfx1250) — matrix operand order repaired and checked by reading, not compiled, not run on hardware
//
#include <hip/hip_runtime.h>
#include <hip/hip_bf16.h>
#include <math.h>

// Problem dims (match reference)
#define T_STEPS 256
#define NENVS   64
#define A_H     16
#define INPUT   12
#define EMB     64
#define EDGE    128
#define RNN     128
#define OUTF    256
#define ROWS    (NENVS * A_H)       // 1024 rows per timestep
#define TNA     (T_STEPS * ROWS)    // 262144 total rows

// Workspace layout (bytes). Requires ~64.3 MB of scratch.
#define X_OFF     0ull                                    // ushort[TNA*128] bf16 x
#define WEDGE_OFF (X_OFF + (size_t)TNA * 128 * 2)         // 64x256 bf16
#define WIH_OFF   (WEDGE_OFF + (size_t)EMB * 256 * 2)     // 384x128 bf16
#define WHH_OFF   (WIH_OFF + (size_t)3 * RNN * RNN * 2)   // 384x128 bf16
#define WOUT_OFF  (WHH_OFF + (size_t)3 * RNN * RNN * 2)   // 256x128 bf16

typedef __attribute__((ext_vector_type(16))) __bf16 v16bf;
typedef __attribute__((ext_vector_type(8)))  float  v8f;
typedef __attribute__((ext_vector_type(4)))  int    v4i;

#ifndef __has_builtin
#define __has_builtin(x) 0
#endif
#if __has_builtin(__builtin_amdgcn_global_load_async_to_lds_b128) && \
    __has_builtin(__builtin_amdgcn_s_wait_asynccnt)
#define HAVE_ASYNC 1
#define ASG __attribute__((address_space(1)))
#define ASL __attribute__((address_space(3)))
#else
#define HAVE_ASYNC 0
#endif

union BFrag {
  v16bf v;
  unsigned short u[16];
};

__device__ __forceinline__ unsigned short bfbits(float f) {
#if defined(__gfx1250__)
  // native RTNE f32->bf16 convert (v_cvt_*bf16_f32) instead of bit twiddling
  union { __bf16 b; unsigned short u; } cv;
  cv.b = (__bf16)f;
  return cv.u;
#else
  union { float f; unsigned u; } in; in.f = f;
  unsigned u = in.u;
  u += 0x7FFFu + ((u >> 16) & 1u);
  return (unsigned short)(u >> 16);
#endif
}

__device__ __forceinline__ void wmma_bf16(v8f& c, const BFrag& a, const BFrag& b) {
  c = __builtin_amdgcn_wmma_f32_16x16x32_bf16(
      /*neg_a=*/false, a.v, /*neg_b=*/false, b.v,
      /*c_mod=*/(short)0, c, /*reuse_a=*/false, /*reuse_b=*/false);
}

// A-fragment (16x32 bf16) from row-major bf16 matrix slice X (row stride ld).
// ISA 16-bit A layout: lanes 0-15: M=lane, K={0..7} in v0..3, {16..23} in v4..7;
// lanes 16-31: M=lane-16, K={8..15} in v0..3, {24..31} in v4..7.
__device__ __forceinline__ void load_a(BFrag& f, const unsigned short* X, int ld, int lane) {
  int m = lane & 15, half = lane >> 4;
#pragma unroll
  for (int v = 0; v < 8; ++v) {
    int kb = (v < 4) ? (half * 8 + v * 2) : (16 + half * 8 + (v - 4) * 2);
    f.u[2 * v]     = X[m * ld + kb];
    f.u[2 * v + 1] = X[m * ld + kb + 1];
  }
}

// B-fragment (32x16 bf16) implementing W^T tile: B[k][n] = W[obase+n][kbase+k],
// W row-major [outF][ld].  ISA 16-bit B layout: N = lane&15, same K order as A:
// lanes 0-15: K={0..7} in v0..3, {16..23} in v4..7; lanes 16-31: K={8..15}, {24..31}.
__device__ __forceinline__ void load_b(BFrag& f, const unsigned short* W, int ld,
                                       int obase, int kbase, int lane) {
  int n = lane & 15, half = lane >> 4;
#pragma unroll
  for (int v = 0; v < 8; ++v) {
    int koff = (v < 4) ? (half * 8 + v * 2) : (16 + half * 8 + (v - 4) * 2);
    f.u[2 * v]     = W[(obase + n) * ld + kbase + koff];
    f.u[2 * v + 1] = W[(obase + n) * ld + kbase + koff + 1];
  }
}

// ---------------------------------------------------------------------------
// Kernel 0: convert weights f32 -> bf16 into workspace (one-time, tiny).
// ---------------------------------------------------------------------------
__global__ void k_convert(const float* __restrict__ We, const float* __restrict__ Wih,
                          const float* __restrict__ Whh, const float* __restrict__ Wout,
                          unsigned short* wEdge, unsigned short* wIh,
                          unsigned short* wHh, unsigned short* wOut) {
  int i = blockIdx.x * blockDim.x + threadIdx.x;
  if (i < EMB * 256)      wEdge[i] = bfbits(We[i]);
  if (i < 3 * RNN * RNN)  { wIh[i] = bfbits(Wih[i]); wHh[i] = bfbits(Whh[i]); }
  if (i < OUTF * RNN)     wOut[i] = bfbits(Wout[i]);
}

// ---------------------------------------------------------------------------
// Kernel 1: x = [relu(pos@Wenc^T+benc), relu(cat(ht,hs)@Wedge^T+bedge)] -> bf16
// 256 threads (8 waves), 128 rows per block; edge GEMM uses bf16 WMMA (K=256).
// W_edge staged to LDS via GLOBAL_LOAD_ASYNC_TO_LDS when available.
// ---------------------------------------------------------------------------
__global__ void k_embed(const float* __restrict__ pos, const float* __restrict__ ht,
                        const float* __restrict__ hs, const float* __restrict__ Wenc,
                        const float* __restrict__ benc, const float* __restrict__ bedge,
                        const unsigned short* __restrict__ WedgeB,
                        unsigned short* __restrict__ x) {
  __shared__ unsigned short sWe[EMB * 256];   // 32 KB bf16 W_edge
  __shared__ float sWenc[EMB * INPUT];
  __shared__ float sbenc[EMB];
  __shared__ float sbedge[EMB];
  __shared__ float spos[128 * INPUT];

  int tid = threadIdx.x;
  int lane = tid & 31, wave = tid >> 5;
  size_t blockRow = (size_t)blockIdx.x * 128;

#if HAVE_ASYNC
#pragma unroll
  for (int j = 0; j < 8; ++j) {            // 2048 x b128 chunks, 8 per thread
    int c = tid + 256 * j;
    __builtin_amdgcn_global_load_async_to_lds_b128(
        (ASG v4i*)(WedgeB + c * 8), (ASL v4i*)(&sWe[c * 8]), 0, 0);
  }
#else
  for (int i = tid; i < EMB * 256; i += 256) sWe[i] = WedgeB[i];
#endif
  for (int i = tid; i < EMB * INPUT; i += 256) sWenc[i] = Wenc[i];
  if (tid < EMB) { sbenc[tid] = benc[tid]; sbedge[tid] = bedge[tid]; }
  for (int i = tid; i < 128 * INPUT; i += 256) spos[i] = pos[blockRow * INPUT + i];
#if HAVE_ASYNC
  __builtin_amdgcn_s_wait_asynccnt(0);
#endif
  __syncthreads();

  // encoder: K=12 too skinny for WMMA -> scalar FMA from LDS
  for (int e = tid; e < 128 * EMB; e += 256) {
    int r = e >> 6, c = e & 63;
    float acc = sbenc[c];
#pragma unroll
    for (int k = 0; k < INPUT; ++k) acc += spos[r * INPUT + k] * sWenc[c * INPUT + k];
    x[(blockRow + r) * 128 + c] = bfbits(fmaxf(acc, 0.f));
  }

  // edge embed: each wave owns 16 rows; 4 N-tiles x 8 K-steps of bf16 WMMA
  size_t rb = blockRow + (size_t)wave * 16;
  v8f acc0 = {}, acc1 = {}, acc2 = {}, acc3 = {};
  int m = lane & 15, half = lane >> 4;
#pragma unroll
  for (int kc = 0; kc < 8; ++kc) {
    BFrag a;
#pragma unroll
    for (int v = 0; v < 8; ++v) {
      int k = kc * 32 + ((v < 4) ? (half * 8 + v * 2) : (16 + half * 8 + (v - 4) * 2));
      const float* src = (k < EDGE) ? (ht + (rb + m) * (size_t)EDGE + k)
                                    : (hs + (rb + m) * (size_t)EDGE + (k - EDGE));
      a.u[2 * v]     = bfbits(src[0]);
      a.u[2 * v + 1] = bfbits(src[1]);
    }
    BFrag b;
    load_b(b, sWe, 256, 0, kc * 32, lane);  wmma_bf16(acc0, a, b);
    load_b(b, sWe, 256, 16, kc * 32, lane); wmma_bf16(acc1, a, b);
    load_b(b, sWe, 256, 32, kc * 32, lane); wmma_bf16(acc2, a, b);
    load_b(b, sWe, 256, 48, kc * 32, lane); wmma_bf16(acc3, a, b);
  }
  int n = lane & 15;
#pragma unroll
  for (int nt = 0; nt < 4; ++nt) {
    v8f acc = (nt == 0) ? acc0 : (nt == 1) ? acc1 : (nt == 2) ? acc2 : acc3;
#pragma unroll
    for (int i = 0; i < 8; ++i) {
      int r = i + 8 * half, c = nt * 16 + n;
      float v = fmaxf(acc[i] + sbedge[c], 0.f);
      x[(rb + r) * 128 + 64 + c] = bfbits(v);
    }
  }
}

// ---------------------------------------------------------------------------
// Kernel 2: GRU scan + output GEMM. One block per env (16 rows of state).
// 512 threads = 16 waves. Weight B-fragments live in registers for all 256
// steps; h / x_t / gi / gh staged in LDS.  x_{t+1} is double-buffered via
// async copy (ASYNCcnt) overlapped with step-t WMMA work when available.
// ---------------------------------------------------------------------------
__global__ void __launch_bounds__(512)
k_scan(const unsigned short* __restrict__ x, const float* __restrict__ h0,
       const float* __restrict__ masks,
       const unsigned short* __restrict__ Wih, const unsigned short* __restrict__ Whh,
       const unsigned short* __restrict__ Wout,
       const float* __restrict__ bih, const float* __restrict__ bhh,
       const float* __restrict__ bout, float* __restrict__ out) {
  __shared__ float          hS[16][RNN];        // f32 hidden state
  __shared__ unsigned short hB[16][RNN];        // bf16 hidden (WMMA A operand)
  __shared__ unsigned short xS[2][16][RNN];     // bf16 x_t tile, double-buffered
  __shared__ float          giS[16][3 * RNN];
  __shared__ float          ghS[16][3 * RNN];
  __shared__ float sbih[3 * RNN], sbhh[3 * RNN], sbout[OUTF];

  int tid = threadIdx.x, lane = tid & 31, wave = tid >> 5;
  int env = blockIdx.x;
  size_t rowbase = (size_t)env * 16;

  for (int i = tid; i < 3 * RNN; i += 512) { sbih[i] = bih[i]; sbhh[i] = bhh[i]; }
  for (int i = tid; i < OUTF; i += 512) sbout[i] = bout[i];
  for (int i = tid; i < 16 * RNN; i += 512) {
    int r = i >> 7, k = i & 127;
    float hv = h0[(rowbase + r) * RNN + k];
    hS[r][k] = hv; hB[r][k] = bfbits(hv);
  }

  // Register-resident weight B-fragments, fixed for all 256 steps.
  // Tiles 0..23 -> gi (W_ih rows), 24..47 -> gh (W_hh rows); wave w owns
  // tiles 3w..3w+2 plus out-tile w (W_out rows 16w..16w+15).
  BFrag bfr[16];
#pragma unroll
  for (int j = 0; j < 3; ++j) {
    int tIdx = wave * 3 + j;
    const unsigned short* W = (tIdx < 24) ? Wih : Whh;
    int ot = (tIdx < 24) ? tIdx : (tIdx - 24);
#pragma unroll
    for (int kc = 0; kc < 4; ++kc) load_b(bfr[j * 4 + kc], W, RNN, ot * 16, kc * 32, lane);
  }
#pragma unroll
  for (int kc = 0; kc < 4; ++kc) load_b(bfr[12 + kc], Wout, RNN, wave * 16, kc * 32, lane);

#if HAVE_ASYNC
  // kick off async copy of x_0 into buffer 0 (16 rows x 128 bf16 = 256 x b128)
  if (tid < 256)
    __builtin_amdgcn_global_load_async_to_lds_b128(
        (ASG v4i*)(x + rowbase * RNN + tid * 8),
        (ASL v4i*)(&xS[0][0][0] + tid * 8), 0, 0);
#endif

  for (int t = 0; t < T_STEPS; ++t) {
    int cur = t & 1;
    __syncthreads();  // prev-step reads of hB / xS[cur^1] complete
    float m = masks[t * NENVS + env];
    // mask reset at segment starts (h *= masks[t])
    for (int i = tid; i < 16 * RNN; i += 512) {
      int r = i >> 7, k = i & 127;
      float hv = hS[r][k] * m;
      hS[r][k] = hv; hB[r][k] = bfbits(hv);
    }
#if HAVE_ASYNC
    __builtin_amdgcn_s_wait_asynccnt(0);       // x_t copy retired (issuing waves)
    __syncthreads();                           // x_t + masked h visible to all
    if (t + 1 < T_STEPS && tid < 256)          // overlap x_{t+1} with step-t WMMA
      __builtin_amdgcn_global_load_async_to_lds_b128(
          (ASG v4i*)(x + ((size_t)(t + 1) * ROWS + rowbase) * RNN + tid * 8),
          (ASL v4i*)(&xS[cur ^ 1][0][0] + tid * 8), 0, 0);
#else
    {
      const unsigned short* xsrc = x + ((size_t)t * ROWS + rowbase) * RNN;
      for (int i = tid; i < 16 * RNN; i += 512) xS[cur][i >> 7][i & 127] = xsrc[i];
      if (t + 1 < T_STEPS)
        __builtin_prefetch(x + ((size_t)(t + 1) * ROWS + rowbase) * RNN + tid * 4, 0, 0);
    }
    __syncthreads();
#endif

    // gi = x_t @ Wih^T, gh = h @ Whh^T : 48 16x16 tiles, 3 per wave, K=128
#pragma unroll
    for (int j = 0; j < 3; ++j) {
      int tIdx = wave * 3 + j;
      bool isGi = tIdx < 24;
      int ot = isGi ? tIdx : (tIdx - 24);
      v8f acc = {};
#pragma unroll
      for (int kc = 0; kc < 4; ++kc) {
        BFrag a;
        load_a(a, (isGi ? &xS[cur][0][0] : &hB[0][0]) + kc * 32, RNN, lane);
        wmma_bf16(acc, a, bfr[j * 4 + kc]);
      }
      int n = lane & 15, half = lane >> 4;
      float* dst = isGi ? &giS[0][0] : &ghS[0][0];
#pragma unroll
      for (int i = 0; i < 8; ++i)
        dst[(i + 8 * half) * (3 * RNN) + ot * 16 + n] = acc[i];
    }
    __syncthreads();

    // GRU elementwise update (torch convention): h = (1-z)*n + z*h
    for (int i = tid; i < 16 * RNN; i += 512) {
      int r = i >> 7, k = i & 127;
      float ir = giS[r][k] + sbih[k],                 hr = ghS[r][k] + sbhh[k];
      float iz = giS[r][RNN + k] + sbih[RNN + k],     hz = ghS[r][RNN + k] + sbhh[RNN + k];
      float in = giS[r][2 * RNN + k] + sbih[2 * RNN + k];
      float hn = ghS[r][2 * RNN + k] + sbhh[2 * RNN + k];
      float rg = 1.f / (1.f + __expf(-(ir + hr)));
      float zg = 1.f / (1.f + __expf(-(iz + hz)));
      float ng = tanhf(in + rg * hn);
      float hv = (1.f - zg) * ng + zg * hS[r][k];
      hS[r][k] = hv; hB[r][k] = bfbits(hv);
    }
    __syncthreads();

    // out_t = h_t @ Wout^T + bout : wave w -> cols [16w,16w+16), K=128
    {
      v8f acc = {};
#pragma unroll
      for (int kc = 0; kc < 4; ++kc) {
        BFrag a;
        load_a(a, &hB[0][0] + kc * 32, RNN, lane);
        wmma_bf16(acc, a, bfr[12 + kc]);
      }
      int n = lane & 15, half = lane >> 4;
#pragma unroll
      for (int i = 0; i < 8; ++i) {
        int r = i + 8 * half, c = wave * 16 + n;
        out[((size_t)t * ROWS + rowbase + r) * OUTF + c] = acc[i] + sbout[c];
      }
    }
  }
  __syncthreads();
  // h_last appended after out in d_out (tuple return order)
  float* hlast = out + (size_t)T_STEPS * ROWS * OUTF;
  for (int i = tid; i < 16 * RNN; i += 512) {
    int r = i >> 7, k = i & 127;
    hlast[(rowbase + r) * RNN + k] = hS[r][k];
  }
}

extern "C" void kernel_launch(void* const* d_in, const int* in_sizes, int n_in,
                              void* d_out, int out_size, void* d_ws, size_t ws_size,
                              hipStream_t stream) {
  const float* pos   = (const float*)d_in[0];
  const float* ht    = (const float*)d_in[1];
  const float* hs    = (const float*)d_in[2];
  const float* h0    = (const float*)d_in[3];
  const float* masks = (const float*)d_in[4];
  const float* Wenc  = (const float*)d_in[5];
  const float* benc  = (const float*)d_in[6];
  const float* Wedge = (const float*)d_in[7];
  const float* bedge = (const float*)d_in[8];
  const float* Wih   = (const float*)d_in[9];
  const float* bih   = (const float*)d_in[10];
  const float* Whh   = (const float*)d_in[11];
  const float* bhh   = (const float*)d_in[12];
  const float* Wout  = (const float*)d_in[13];
  const float* bout  = (const float*)d_in[14];
  float* out = (float*)d_out;

  char* ws = (char*)d_ws;
  unsigned short* xw    = (unsigned short*)(ws + X_OFF);
  unsigned short* wEdge = (unsigned short*)(ws + WEDGE_OFF);
  unsigned short* wIh   = (unsigned short*)(ws + WIH_OFF);
  unsigned short* wHh   = (unsigned short*)(ws + WHH_OFF);
  unsigned short* wOut  = (unsigned short*)(ws + WOUT_OFF);

  (void)in_sizes; (void)n_in; (void)out_size; (void)ws_size;

  k_convert<<<192, 256, 0, stream>>>(Wedge, Wih, Whh, Wout, wEdge, wIh, wHh, wOut);
  k_embed<<<TNA / 128, 256, 0, stream>>>(pos, ht, hs, Wenc, benc, bedge, wEdge, xw);
  k_scan<<<NENVS, 512, 0, stream>>>(xw, h0, masks, wIh, wHh, wOut, bih, bhh, bout, out);
}
